// GATv2Model_4922032521307
// MI455X (gfx1250) — matrix operand order repaired and checked by reading, not compiled, not run on hardware
//
#include <hip/hip_runtime.h>
#include <hip/hip_bf16.h>

// GATv2 (2 layers) for MI455X / gfx1250, wave32.
// Dense transforms via v_wmma_f32_16x16x32_bf16 (16x64 tile per wave, 4 accs);
// edge softmax/aggregation via L2-resident float atomics.

#define IN_DIM   512
#define HID_DIM  128
#define NHEADS   2
#define HC1      (NHEADS * HID_DIM)   // 256
#define OUT_DIM  64

typedef __attribute__((ext_vector_type(16))) __bf16 v16bf;
typedef __attribute__((ext_vector_type(8)))  __bf16 v8bf;
typedef __attribute__((ext_vector_type(8)))  float  v8f;

// ---- float <-> order-preserving uint (for atomicMax on floats) ----
static __device__ __forceinline__ unsigned f2ord(float f) {
  unsigned u = __float_as_uint(f);
  return (u & 0x80000000u) ? ~u : (u | 0x80000000u);
}
static __device__ __forceinline__ float ord2f(unsigned u) {
  return (u & 0x80000000u) ? __uint_as_float(u & 0x7fffffffu)
                           : __uint_as_float(~u);
}
#define ORD_NEG_INF 0x007FFFFFu   // f2ord(-inf)

// ------------------------- small utility kernels -------------------------
__global__ void k_fill_f32(float* __restrict__ p, float v, long long n) {
  long long i = (long long)blockIdx.x * blockDim.x + threadIdx.x;
  if (i < n) p[i] = v;
}
__global__ void k_fill_u32(unsigned* __restrict__ p, unsigned v, long long n) {
  long long i = (long long)blockIdx.x * blockDim.x + threadIdx.x;
  if (i < n) p[i] = v;
}
__global__ void k_f32_to_bf16(const float* __restrict__ s, __bf16* __restrict__ d,
                              long long n) {
  long long i = (long long)blockIdx.x * blockDim.x + threadIdx.x;
  if (i < n) d[i] = (__bf16)s[i];
}
// W[K,N] (f32, row-major) -> Wt[N,K] (bf16) so B fragments load contiguously in K
__global__ void k_transpose_bf16(const float* __restrict__ W, __bf16* __restrict__ Wt,
                                 int K, int N) {
  long long i = (long long)blockIdx.x * blockDim.x + threadIdx.x;
  if (i >= (long long)K * N) return;
  int k = (int)(i / N), n = (int)(i % N);
  Wt[(size_t)n * K + k] = (__bf16)W[i];
}
__global__ void k_elu_to_bf16(const float* __restrict__ s, __bf16* __restrict__ d,
                              long long n) {
  long long i = (long long)blockIdx.x * blockDim.x + threadIdx.x;
  if (i >= n) return;
  float x = s[i];
  float v = (x > 0.f) ? x : (__expf(x) - 1.f);
  d[i] = (__bf16)v;
}
__global__ void k_bias_init(float* __restrict__ out, const float* __restrict__ bias,
                            long long n, int HCv) {
  long long i = (long long)blockIdx.x * blockDim.x + threadIdx.x;
  if (i < n) out[i] = bias[(int)(i % HCv)];
}

// -------- WMMA GEMM: C[M,N] = A(bf16)[M,K] @ Bt(bf16)[N,K]^T + bias --------
// One wave per 16x64 output tile (4 accumulators): A fragment is loaded once
// per k-step and reused across 4 WMMAs (independent accumulators -> no D->A/B
// hazard stalls). M mult of 16, N mult of 64, K mult of 32 (holds here).
__global__ void k_gemm_bf16_wmma(const __bf16* __restrict__ A,
                                 const __bf16* __restrict__ Bt,
                                 const float* __restrict__ bias,
                                 float* __restrict__ C,
                                 int M, int N, int K) {
  const int lane = threadIdx.x & 31;
  const int wave = blockIdx.x * (blockDim.x >> 5) + (threadIdx.x >> 5);
  const int tilesN = N >> 6;             // 64-wide N tiles
  const int totalTiles = (M >> 4) * tilesN;
  if (wave >= totalTiles) return;        // whole-wave uniform: EXEC stays all-1s
  const int tm = wave / tilesN;
  const int tn = wave - tm * tilesN;

  const int hi   = lane >> 4;            // 0: lanes 0-15, 1: lanes 16-31
  const int rA   = (tm << 4) + (lane & 15);
  const size_t aRow = (size_t)rA * K;
  size_t bRow[4];
#pragma unroll
  for (int j = 0; j < 4; ++j)
    bRow[j] = (size_t)((tn << 6) + (j << 4) + (lane & 15)) * K;

  v8f acc[4] = {};
  for (int kk = 0; kk < K; kk += 32) {
    // A (16x32 bf16): lane holds 16 contiguous K values [kA, kA+16)
    const int kA = kk + (hi << 4);
    v8bf a0 = *(const v8bf*)(A + aRow + kA);
    v8bf a1 = *(const v8bf*)(A + aRow + kA + 8);
    v16bf av = __builtin_shufflevector(a0, a1, 0, 1, 2, 3, 4, 5, 6, 7,
                                               8, 9, 10, 11, 12, 13, 14, 15);
    // B (32x16 bf16): lane holds 16 contiguous K values of its column
    const int kB = kk + (hi << 4);
#pragma unroll
    for (int j = 0; j < 4; ++j) {
      v16bf bv = *(const v16bf*)(Bt + bRow[j] + kB);
      acc[j] = __builtin_amdgcn_wmma_f32_16x16x32_bf16(
          /*neg_a=*/false, av, /*neg_b=*/false, bv,
          /*c_mod=*/(short)0, acc[j], /*reuse_a=*/false, /*reuse_b=*/false);
    }
  }
  // C/D layout: VGPR v -> row v (lanes 0-15) / v+8 (lanes 16-31); col = lane%16
  const int row0 = (tm << 4) + (hi << 3);
#pragma unroll
  for (int j = 0; j < 4; ++j) {
    const int col = (tn << 6) + (j << 4) + (lane & 15);
    const float bb = bias[col];
#pragma unroll
    for (int v = 0; v < 8; ++v)
      C[(size_t)(row0 + v) * N + col] = acc[j][v] + bb;
  }
}

// ------------------------- graph / edge kernels -------------------------
__global__ void k_deg_asum(const int* __restrict__ dst, const float* __restrict__ eattr,
                           float* __restrict__ deg, float* __restrict__ asum, int E) {
  int i = blockIdx.x * blockDim.x + threadIdx.x;
  if (i >= E) return;
  int d = dst[i];
  atomicAdd(&deg[d], 1.f);
  atomicAdd(&asum[d], eattr[i]);
}
__global__ void k_loop_attr(const float* __restrict__ deg, const float* __restrict__ asum,
                            float* __restrict__ lattr, int Nn) {
  int i = blockIdx.x * blockDim.x + threadIdx.x;
  if (i < Nn) lattr[i] = asum[i] / fmaxf(deg[i], 1.f);
}

// Pass 1: wave per edge. alpha[e,h] = att_h . leaky_relu(xl[s]+xr[d]+ea*We)
__global__ void k_edge_alpha(const float* __restrict__ xl, const float* __restrict__ xr,
                             const int* __restrict__ src, const int* __restrict__ dst,
                             const float* __restrict__ eattr, const float* __restrict__ lattr,
                             const float* __restrict__ We, const float* __restrict__ att,
                             float* __restrict__ alpha, unsigned* __restrict__ amax,
                             int E, int Nn, int H, int C) {
  const int lane = threadIdx.x & 31;
  const int e = blockIdx.x * (blockDim.x >> 5) + (threadIdx.x >> 5);
  if (e >= E + Nn) return;
  int s, d; float ea;
  if (e < E) { s = src[e]; d = dst[e]; ea = eattr[e]; }
  else       { s = e - E; d = s;       ea = lattr[s]; }
  const int HCv = H * C;
  const int CPL = HCv >> 5;              // channels per lane (C % CPL == 0)
  const int ch0 = lane * CPL;
  const int h   = ch0 / C;
  const float* xls = xl + (size_t)s * HCv;
  const float* xrd = xr + (size_t)d * HCv;
  __builtin_prefetch(xls, 0, 0);         // global_prefetch_b8
  __builtin_prefetch(xrd, 0, 0);
  float acc = 0.f;
  for (int i = 0; i < CPL; ++i) {
    const int ch = ch0 + i;
    float m = xls[ch] + xrd[ch] + ea * We[ch];
    m = (m > 0.f) ? m : 0.2f * m;        // leaky_relu(0.2)
    acc += m * att[ch];                  // att flat index == ch (h*C + c)
  }
  const int LPH = 32 / H;                // lanes per head
  for (int off = LPH >> 1; off > 0; off >>= 1)
    acc += __shfl_xor(acc, off, LPH);
  if ((lane & (LPH - 1)) == 0) {
    alpha[(size_t)e * H + h] = acc;
    atomicMax(&amax[(size_t)d * H + h], f2ord(acc));
  }
}

// Pass 2: thread per (edge, head): ez = exp(alpha - amax[d]); denom += ez
__global__ void k_edge_expsum(const float* __restrict__ alpha, const unsigned* __restrict__ amax,
                              const int* __restrict__ dst,
                              float* __restrict__ ez, float* __restrict__ denom,
                              int E, int Nn, int H) {
  long long i = (long long)blockIdx.x * blockDim.x + threadIdx.x;
  long long tot = (long long)(E + Nn) * H;
  if (i >= tot) return;
  int e = (int)(i / H), h = (int)(i % H);
  int d = (e < E) ? dst[e] : (e - E);
  float v = __expf(alpha[i] - ord2f(amax[(size_t)d * H + h]));
  ez[i] = v;
  atomicAdd(&denom[(size_t)d * H + h], v);
}

// Pass 3: wave per edge: out[d] += xl[s] * ez/denom  (bias pre-seeded in out)
__global__ void k_edge_scatter(const float* __restrict__ xl,
                               const int* __restrict__ src, const int* __restrict__ dst,
                               const float* __restrict__ ez, const float* __restrict__ denom,
                               float* __restrict__ out,
                               int E, int Nn, int H, int C) {
  const int lane = threadIdx.x & 31;
  const int e = blockIdx.x * (blockDim.x >> 5) + (threadIdx.x >> 5);
  if (e >= E + Nn) return;
  int s, d;
  if (e < E) { s = src[e]; d = dst[e]; } else { s = e - E; d = s; }
  const int HCv = H * C;
  const float* xls = xl + (size_t)s * HCv;
  float* od = out + (size_t)d * HCv;
  __builtin_prefetch(xls, 0, 0);
  for (int ch = lane; ch < HCv; ch += 32) {
    int h = ch / C;
    float a = ez[(size_t)e * H + h] / denom[(size_t)d * H + h];
    atomicAdd(&od[ch], xls[ch] * a);
  }
}

// ------------------------------ launcher ------------------------------
extern "C" void kernel_launch(void* const* d_in, const int* in_sizes, int n_in,
                              void* d_out, int out_size, void* d_ws, size_t ws_size,
                              hipStream_t stream) {
  const float* x     = (const float*)d_in[0];
  const int*   eidx  = (const int*)d_in[1];
  const float* eattr = (const float*)d_in[2];
  const float* W1l   = (const float*)d_in[3];
  const float* b1l   = (const float*)d_in[4];
  const float* W1r   = (const float*)d_in[5];
  const float* b1r   = (const float*)d_in[6];
  const float* W1e   = (const float*)d_in[7];
  const float* att1  = (const float*)d_in[8];
  const float* bias1 = (const float*)d_in[9];
  const float* W2l   = (const float*)d_in[10];
  const float* b2l   = (const float*)d_in[11];
  const float* W2r   = (const float*)d_in[12];
  const float* b2r   = (const float*)d_in[13];
  const float* W2e   = (const float*)d_in[14];
  const float* att2  = (const float*)d_in[15];
  const float* bias2 = (const float*)d_in[16];

  const int Nn = in_sizes[0] / IN_DIM;    // 20000
  const int E  = in_sizes[1] / 2;         // 320000
  const int* src  = eidx;
  const int* dstI = eidx + E;
  const int ET = E + Nn;                  // edges incl. self-loops

  // ---- carve workspace ----
  char* base = (char*)d_ws;
  size_t off = 0;
  auto carve = [&](size_t bytes) -> void* {
    void* p = base + off;
    off = (off + bytes + 255) & ~(size_t)255;
    return p;
  };
  __bf16* xbf   = (__bf16*)carve((size_t)Nn * IN_DIM * sizeof(__bf16));
  __bf16* w1lt  = (__bf16*)carve((size_t)IN_DIM * HC1 * sizeof(__bf16));
  __bf16* w1rt  = (__bf16*)carve((size_t)IN_DIM * HC1 * sizeof(__bf16));
  __bf16* w2lt  = (__bf16*)carve((size_t)HC1 * OUT_DIM * sizeof(__bf16));
  __bf16* w2rt  = (__bf16*)carve((size_t)HC1 * OUT_DIM * sizeof(__bf16));
  float*  XL1   = (float*)carve((size_t)Nn * HC1 * sizeof(float));
  float*  XR1   = (float*)carve((size_t)Nn * HC1 * sizeof(float));
  float*  deg   = (float*)carve((size_t)Nn * sizeof(float));
  float*  asum  = (float*)carve((size_t)Nn * sizeof(float));
  float*  lattr = (float*)carve((size_t)Nn * sizeof(float));
  float*  al1   = (float*)carve((size_t)ET * NHEADS * sizeof(float));
  float*  ez1   = (float*)carve((size_t)ET * NHEADS * sizeof(float));
  unsigned* am1 = (unsigned*)carve((size_t)Nn * NHEADS * sizeof(unsigned));
  float*  den1  = (float*)carve((size_t)Nn * NHEADS * sizeof(float));
  float*  out1  = (float*)carve((size_t)Nn * HC1 * sizeof(float));
  __bf16* hbf   = (__bf16*)carve((size_t)Nn * HC1 * sizeof(__bf16));
  float*  XL2   = (float*)carve((size_t)Nn * OUT_DIM * sizeof(float));
  float*  XR2   = (float*)carve((size_t)Nn * OUT_DIM * sizeof(float));
  float*  al2   = (float*)carve((size_t)ET * sizeof(float));
  float*  ez2   = (float*)carve((size_t)ET * sizeof(float));
  unsigned* am2 = (unsigned*)carve((size_t)Nn * sizeof(unsigned));
  float*  den2  = (float*)carve((size_t)Nn * sizeof(float));
  (void)ws_size;

  const int T = 256;                       // 8 waves per block
  auto blk1d = [&](long long n) { return (unsigned)((n + T - 1) / T); };
  auto blkW  = [&](long long waves) { return (unsigned)((waves + 7) / 8); };

  // ---- bf16 conversions / weight transposes ----
  k_f32_to_bf16<<<blk1d((long long)Nn * IN_DIM), T, 0, stream>>>(x, xbf, (long long)Nn * IN_DIM);
  k_transpose_bf16<<<blk1d((long long)IN_DIM * HC1), T, 0, stream>>>(W1l, w1lt, IN_DIM, HC1);
  k_transpose_bf16<<<blk1d((long long)IN_DIM * HC1), T, 0, stream>>>(W1r, w1rt, IN_DIM, HC1);
  k_transpose_bf16<<<blk1d((long long)HC1 * OUT_DIM), T, 0, stream>>>(W2l, w2lt, HC1, OUT_DIM);
  k_transpose_bf16<<<blk1d((long long)HC1 * OUT_DIM), T, 0, stream>>>(W2r, w2rt, HC1, OUT_DIM);

  // ---- layer 1 dense: XL1 = x@W1l+b1l, XR1 = x@W1r+b1r ----
  {
    long long tiles = (long long)(Nn / 16) * (HC1 / 64);
    k_gemm_bf16_wmma<<<blkW(tiles), T, 0, stream>>>(xbf, w1lt, b1l, XL1, Nn, HC1, IN_DIM);
    k_gemm_bf16_wmma<<<blkW(tiles), T, 0, stream>>>(xbf, w1rt, b1r, XR1, Nn, HC1, IN_DIM);
  }

  // ---- self-loop edge attr (mean of incoming) ----
  k_fill_f32<<<blk1d(Nn), T, 0, stream>>>(deg, 0.f, Nn);
  k_fill_f32<<<blk1d(Nn), T, 0, stream>>>(asum, 0.f, Nn);
  k_deg_asum<<<blk1d(E), T, 0, stream>>>(dstI, eattr, deg, asum, E);
  k_loop_attr<<<blk1d(Nn), T, 0, stream>>>(deg, asum, lattr, Nn);

  // ---- layer 1 segment softmax + aggregate ----
  k_fill_u32<<<blk1d((long long)Nn * NHEADS), T, 0, stream>>>(am1, ORD_NEG_INF, (long long)Nn * NHEADS);
  k_fill_f32<<<blk1d((long long)Nn * NHEADS), T, 0, stream>>>(den1, 0.f, (long long)Nn * NHEADS);
  k_edge_alpha<<<blkW(ET), T, 0, stream>>>(XL1, XR1, src, dstI, eattr, lattr,
                                           W1e, att1, al1, am1, E, Nn, NHEADS, HID_DIM);
  k_edge_expsum<<<blk1d((long long)ET * NHEADS), T, 0, stream>>>(al1, am1, dstI, ez1, den1, E, Nn, NHEADS);
  k_bias_init<<<blk1d((long long)Nn * HC1), T, 0, stream>>>(out1, bias1, (long long)Nn * HC1, HC1);
  k_edge_scatter<<<blkW(ET), T, 0, stream>>>(XL1, src, dstI, ez1, den1, out1, E, Nn, NHEADS, HID_DIM);

  // ---- ELU -> bf16 activation for layer 2 ----
  k_elu_to_bf16<<<blk1d((long long)Nn * HC1), T, 0, stream>>>(out1, hbf, (long long)Nn * HC1);

  // ---- layer 2 dense: XL2 = h@W2l+b2l, XR2 = h@W2r+b2r ----
  {
    long long tiles = (long long)(Nn / 16) * (OUT_DIM / 64);
    k_gemm_bf16_wmma<<<blkW(tiles), T, 0, stream>>>(hbf, w2lt, b2l, XL2, Nn, OUT_DIM, HC1);
    k_gemm_bf16_wmma<<<blkW(tiles), T, 0, stream>>>(hbf, w2rt, b2r, XR2, Nn, OUT_DIM, HC1);
  }

  // ---- layer 2 segment softmax + aggregate straight into d_out ----
  float* out2 = (float*)d_out;
  k_fill_u32<<<blk1d(Nn), T, 0, stream>>>(am2, ORD_NEG_INF, Nn);
  k_fill_f32<<<blk1d(Nn), T, 0, stream>>>(den2, 0.f, Nn);
  k_edge_alpha<<<blkW(ET), T, 0, stream>>>(XL2, XR2, src, dstI, eattr, lattr,
                                           W2e, att2, al2, am2, E, Nn, 1, OUT_DIM);
  k_edge_expsum<<<blk1d(ET), T, 0, stream>>>(al2, am2, dstI, ez2, den2, E, Nn, 1);
  k_bias_init<<<blk1d((long long)Nn * OUT_DIM), T, 0, stream>>>(out2, bias2, (long long)Nn * OUT_DIM, OUT_DIM);
  k_edge_scatter<<<blkW(ET), T, 0, stream>>>(XL2, src, dstI, ez2, den2, out2, E, Nn, 1, OUT_DIM);
}
